// TransformerTranslator_43482248905101
// MI455X (gfx1250) — compile-verified, matrix operand order checked by reading, not run on hardware
//
#include <hip/hip_runtime.h>
#include <hip/hip_bf16.h>

typedef __bf16 bf16;
typedef __attribute__((ext_vector_type(16))) __bf16 v16bf;
typedef __attribute__((ext_vector_type(8)))  float  v8f;
typedef __attribute__((ext_vector_type(4)))  unsigned int u32x4;

union F16frag { v16bf v; u32x4 q[2]; };

#define TT   32
#define HH   768
#define DKV  96
#define DFF  2048
#define X_LD 776     // 32x776 bf16 = 49664B, conflict-free row stride, 16B aligned
#define QK_LD 104    // q/k/attn tiles
#define F_LD 72      // FFN f-chunk

struct TxParams {
  const int* inputs;
  const float* emb; const float* pos;
  const float* bq1; const float* bk1; const float* bv1;
  const float* bq2; const float* bk2; const float* bv2;
  const float* bp;  const float* g1;  const float* be1;
  const float* bf1; const float* bf2; const float* g2; const float* be2; const float* bo;
  const bf16* wqkv; const bf16* wp; const bf16* w1; const bf16* w2; const bf16* wo;
  float* out;
};

static __device__ __forceinline__ v8f vzero() {
  v8f z = {0.f,0.f,0.f,0.f,0.f,0.f,0.f,0.f};
  return z;
}

static __device__ __forceinline__ v8f wmma_bf16(v16bf a, v16bf b, v8f c) {
  // D = A(16x32 bf16) * B(32x16 bf16) + C(16x16 f32)
  return __builtin_amdgcn_wmma_f32_16x16x32_bf16(false, a, false, b, (short)0, c, false, false);
}

// A-fragment and B-fragment share addressing: "row" r0+lane&15, two 16B chunks
// at k0 + (lane>>4)*8 and +16.  For B this assumes a row-major (N,K) matrix
// (weight layout), which equals column-of-B.
static __device__ __forceinline__ v16bf ld_frag(const bf16* base, int ld, int r0, int lane, int k0) {
  int r = r0 + (lane & 15);
  int k = k0 + ((lane >> 4) << 3);
  const bf16* p = base + r * ld + k;
  F16frag f;
  f.q[0] = *(const u32x4*)(p);
  f.q[1] = *(const u32x4*)(p + 16);
  return f.v;
}

static __device__ __forceinline__ v8f gemm_k(const bf16* A, int lda, int m0,
                                             const bf16* Bw, int ldb, int n0,
                                             int K, int lane, v8f acc) {
  for (int k0 = 0; k0 < K; k0 += 32) {
    v16bf a = ld_frag(A, lda, m0, lane, k0);
    v16bf b = ld_frag(Bw, ldb, n0, lane, k0);
    acc = wmma_bf16(a, b, acc);
  }
  return acc;
}

// in-place LayerNorm over rows of sX (32 rows x 768), 8 lanes per row
static __device__ __forceinline__ void layernorm_rows(bf16* X, const float* g, const float* be, int tid) {
  int row = tid >> 3, sub = tid & 7;
  bf16* xr = X + row * X_LD + sub * 96;
  float s = 0.f, ss = 0.f;
  #pragma unroll 8
  for (int c = 0; c < 96; ++c) { float v = (float)xr[c]; s += v; ss += v * v; }
  s += __shfl_xor(s, 1, 8); ss += __shfl_xor(ss, 1, 8);
  s += __shfl_xor(s, 2, 8); ss += __shfl_xor(ss, 2, 8);
  s += __shfl_xor(s, 4, 8); ss += __shfl_xor(ss, 4, 8);
  float mean = s * (1.f / 768.f);
  float var  = ss * (1.f / 768.f) - mean * mean;
  float rstd = rsqrtf(var + 1e-5f);
  const float* gp = g + sub * 96;
  const float* bp = be + sub * 96;
  #pragma unroll 8
  for (int c = 0; c < 96; ++c) {
    float v = ((float)xr[c] - mean) * rstd * gp[c] + bp[c];
    xr[c] = (bf16)v;
  }
}

__global__ void conv_f32_bf16(const float* __restrict__ s, bf16* __restrict__ d, int n) {
  int i = blockIdx.x * 256 + threadIdx.x;
  if (i < n) d[i] = (bf16)s[i];
}

__global__ __launch_bounds__(256) void tx_main(TxParams p) {
  // LDS: x/h/o (32 x 768 bf16, padded) + scratch regions
  __shared__ __align__(16) bf16 sX[TT * X_LD];                 // 49664 B
  __shared__ __align__(16) unsigned char sS[15360];            // 15360 B
  bf16*  R1 = (bf16*)sS;              // q (32xQK_LD) | scores f32 | attn-out | f-chunk
  bf16*  R2 = (bf16*)(sS + 6656);     // k (32xQK_LD) | vT (96x32)
  bf16*  R3 = (bf16*)(sS + 13312);    // softmax probs bf16 (32x32)
  float* sf = (float*)sS;             // scores f32 view of R1

  const int tid  = threadIdx.x;
  const int wave = tid >> 5;
  const int lane = tid & 31;
  const int col16 = lane & 15;
  const int half  = lane >> 4;
  const int n = blockIdx.x;

  // ---- embed: x = emb[inputs] + pos (f32 -> bf16 into LDS)
  for (int t = wave; t < TT; t += 8) {
    int idx = p.inputs[n * TT + t];
    const float* er = p.emb + (size_t)idx * HH;
    const float* pr = p.pos + t * HH;
    bf16* xr = sX + t * X_LD;
    for (int c = lane; c < HH; c += 32) xr[c] = (bf16)(er[c] + pr[c]);
  }
  __syncthreads();

  // persistent accumulators: wave w owns N-tiles [6w, 6w+6) x 2 M-tiles of the
  // (32 x 768) output of the Wp projection (then reused for the FFN output)
  v8f acc12[12];
  #pragma unroll
  for (int t = 0; t < 12; ++t) acc12[t] = vzero();

  // ---- two attention heads
  for (int h = 0; h < 2; ++h) {
    const bf16* wq = p.wqkv + (size_t)(h * 3 + 0) * DKV * HH;
    const bf16* wk = wq + (size_t)DKV * HH;
    const bf16* wv = wk + (size_t)DKV * HH;
    const float* bq = h ? p.bq2 : p.bq1;
    const float* bk = h ? p.bk2 : p.bk1;
    const float* bv = h ? p.bv2 : p.bv1;

    // q -> R1, k -> R2   (24 tiles of 16x16, K = 768)
    for (int t = wave; t < 24; t += 8) {
      int isK = (t >= 12);
      int tt  = isK ? t - 12 : t;
      int mt = tt & 1, nt = tt >> 1;
      v8f acc = gemm_k(sX, X_LD, mt * 16, isK ? wk : wq, HH, nt * 16, HH, lane, vzero());
      const float* bias = isK ? bk : bq;
      bf16* dst = isK ? R2 : R1;
      int cc = nt * 16 + col16;
      float b = bias[cc];
      #pragma unroll
      for (int r = 0; r < 8; ++r)
        dst[(mt * 16 + r + 8 * half) * QK_LD + cc] = (bf16)(acc[r] + b);
    }
    __syncthreads();

    // scores = q @ k^T   (4 tiles, K = 96) -- keep in registers across barrier
    v8f sacc = vzero();
    if (wave < 4) {
      int mt = wave & 1, nt = wave >> 1;
      sacc = gemm_k(R1, QK_LD, mt * 16, R2, QK_LD, nt * 16, DKV, lane, vzero());
    }
    __syncthreads();

    // store scores f32 into R1 (q dead); compute vT into R2 (k dead)
    if (wave < 4) {
      int mt = wave & 1, nt = wave >> 1;
      int cc = nt * 16 + col16;
      #pragma unroll
      for (int r = 0; r < 8; ++r)
        sf[(mt * 16 + r + 8 * half) * 32 + cc] = sacc[r];
    }
    for (int t = wave; t < 12; t += 8) {
      int mt = t & 1, nt = t >> 1;
      v8f acc = gemm_k(sX, X_LD, mt * 16, wv, HH, nt * 16, HH, lane, vzero());
      int d = nt * 16 + col16;
      float b = bv[d];
      __align__(16) bf16 tmp[8];
      #pragma unroll
      for (int r = 0; r < 8; ++r) tmp[r] = (bf16)(acc[r] + b);
      *(u32x4*)(R2 + d * 32 + mt * 16 + 8 * half) = *(const u32x4*)tmp;  // vT[d][t]
    }
    __syncthreads();

    // softmax rows (wave 0, one row per lane), write bf16 probs to R3
    if (wave == 0) {
      float vv[32];
      float m = -3.4e38f;
      #pragma unroll
      for (int j = 0; j < 32; ++j) { vv[j] = sf[lane * 32 + j] * 0.1020620726f; m = fmaxf(m, vv[j]); }
      float s = 0.f;
      #pragma unroll
      for (int j = 0; j < 32; ++j) { vv[j] = __expf(vv[j] - m); s += vv[j]; }
      float inv = 1.f / s;
      #pragma unroll
      for (int j = 0; j < 32; ++j) R3[lane * 32 + j] = (bf16)(vv[j] * inv);
    }
    __syncthreads();

    // attn = probs @ v  (12 tiles, K = 32) -> R1
    for (int t = wave; t < 12; t += 8) {
      int mt = t & 1, nt = t >> 1;
      v8f acc = gemm_k(R3, 32, mt * 16, R2, 32, nt * 16, 32, lane, vzero());
      int cc = nt * 16 + col16;
      #pragma unroll
      for (int r = 0; r < 8; ++r)
        R1[(mt * 16 + r + 8 * half) * QK_LD + cc] = (bf16)acc[r];
    }
    __syncthreads();

    // projection accumulate: acc12 += attn @ Wp[:, h*96 : h*96+96]^T
    #pragma unroll
    for (int t = 0; t < 12; ++t) {
      int mt = t & 1, ntile = wave * 6 + (t >> 1);
      acc12[t] = gemm_k(R1, QK_LD, mt * 16, p.wp + h * DKV, 2 * DKV, ntile * 16, DKV, lane, acc12[t]);
    }
    __syncthreads();
  }

  // ---- residual + LN1 (in place over sX)
  #pragma unroll
  for (int t = 0; t < 12; ++t) {
    int mt = t & 1, ntile = wave * 6 + (t >> 1);
    int cc = ntile * 16 + col16;
    float b = p.bp[cc];
    #pragma unroll
    for (int r = 0; r < 8; ++r) {
      int row = mt * 16 + r + 8 * half;
      float v = acc12[t][r] + b + (float)sX[row * X_LD + cc];
      sX[row * X_LD + cc] = (bf16)v;
    }
  }
  __syncthreads();
  layernorm_rows(sX, p.g1, p.be1, tid);
  __syncthreads();

  // ---- FFN: loop DF in chunks of 64; f-chunk in LDS, output in registers
  #pragma unroll
  for (int t = 0; t < 12; ++t) acc12[t] = vzero();
  for (int ch = 0; ch < 32; ++ch) {
    int df0 = ch * 64;
    { // f = relu(h @ W1_chunk^T + b1) : 8 tiles, one per wave, K = 768
      int mt = wave & 1, nt = wave >> 1;
      v8f acc = gemm_k(sX, X_LD, mt * 16, p.w1 + (size_t)df0 * HH, HH, nt * 16, HH, lane, vzero());
      int cc = nt * 16 + col16;
      float b = p.bf1[df0 + cc];
      #pragma unroll
      for (int r = 0; r < 8; ++r)
        R1[(mt * 16 + r + 8 * half) * F_LD + cc] = (bf16)fmaxf(acc[r] + b, 0.f);
    }
    __syncthreads();
    // acc12 += f_chunk @ W2[:, df0:df0+64]^T  (K = 64)
    #pragma unroll
    for (int t = 0; t < 12; ++t) {
      int mt = t & 1, ntile = wave * 6 + (t >> 1);
      acc12[t] = gemm_k(R1, F_LD, mt * 16, p.w2 + df0, DFF, ntile * 16, 64, lane, acc12[t]);
    }
    __syncthreads();
  }

  // ---- residual + LN2
  #pragma unroll
  for (int t = 0; t < 12; ++t) {
    int mt = t & 1, ntile = wave * 6 + (t >> 1);
    int cc = ntile * 16 + col16;
    float b = p.bf2[cc];
    #pragma unroll
    for (int r = 0; r < 8; ++r) {
      int row = mt * 16 + r + 8 * half;
      float v = acc12[t][r] + b + (float)sX[row * X_LD + cc];
      sX[row * X_LD + cc] = (bf16)v;
    }
  }
  __syncthreads();
  layernorm_rows(sX, p.g2, p.be2, tid);
  __syncthreads();

  // ---- output: (32 x 2) = o @ Wo^T + bo
  if (tid < 64) {
    int row = tid >> 1, oc = tid & 1;
    const bf16* xr = sX + row * X_LD;
    const bf16* wr = p.wo + oc * HH;
    float acc = p.bo[oc];
    for (int c = 0; c < HH; ++c) acc += (float)xr[c] * (float)wr[c];
    p.out[((size_t)n * TT + row) * 2 + oc] = acc;
  }
}

extern "C" void kernel_launch(void* const* d_in, const int* in_sizes, int n_in,
                              void* d_out, int out_size, void* d_ws, size_t ws_size,
                              hipStream_t stream) {
  (void)in_sizes; (void)n_in; (void)out_size; (void)ws_size;
  bf16* ws = (bf16*)d_ws;

  // bf16 weight arena layout (element offsets)
  const size_t OFF_WQKV = 0;                 // q1,k1,v1,q2,k2,v2 : 6 * 96*768
  const size_t OFF_WP   = 442368;            // 768*192
  const size_t OFF_W1   = 589824;            // 2048*768
  const size_t OFF_W2   = 2162688;           // 768*2048
  const size_t OFF_WO   = 3735552;           // 2*768

  struct Seg { int src; size_t off; int n; };
  const Seg segs[10] = {
    { 7, OFF_WQKV + 0 * 73728, 73728 },   // Wq1
    { 3, OFF_WQKV + 1 * 73728, 73728 },   // Wk1
    { 5, OFF_WQKV + 2 * 73728, 73728 },   // Wv1
    { 13, OFF_WQKV + 3 * 73728, 73728 },  // Wq2
    { 9, OFF_WQKV + 4 * 73728, 73728 },   // Wk2
    { 11, OFF_WQKV + 5 * 73728, 73728 },  // Wv2
    { 15, OFF_WP, 147456 },               // Wp
    { 19, OFF_W1, 1572864 },              // W1
    { 21, OFF_W2, 1572864 },              // W2
    { 25, OFF_WO, 1536 },                 // Wo
  };
  for (int i = 0; i < 10; ++i) {
    conv_f32_bf16<<<dim3((segs[i].n + 255) / 256), dim3(256), 0, stream>>>(
        (const float*)d_in[segs[i].src], ws + segs[i].off, segs[i].n);
  }

  TxParams p;
  p.inputs = (const int*)d_in[0];
  p.emb = (const float*)d_in[1];  p.pos = (const float*)d_in[2];
  p.bq1 = (const float*)d_in[8];  p.bk1 = (const float*)d_in[4];  p.bv1 = (const float*)d_in[6];
  p.bq2 = (const float*)d_in[14]; p.bk2 = (const float*)d_in[10]; p.bv2 = (const float*)d_in[12];
  p.bp  = (const float*)d_in[16]; p.g1  = (const float*)d_in[17]; p.be1 = (const float*)d_in[18];
  p.bf1 = (const float*)d_in[20]; p.bf2 = (const float*)d_in[22];
  p.g2  = (const float*)d_in[23]; p.be2 = (const float*)d_in[24]; p.bo = (const float*)d_in[26];
  p.wqkv = ws + OFF_WQKV; p.wp = ws + OFF_WP; p.w1 = ws + OFF_W1;
  p.w2 = ws + OFF_W2; p.wo = ws + OFF_WO;
  p.out = (float*)d_out;

  tx_main<<<dim3(4096), dim3(256), 0, stream>>>(p);
}
